// GRU_Layer_70300024701421
// MI455X (gfx1250) — matrix operand order repaired and checked by reading, not compiled, not run on hardware
//
#include <hip/hip_runtime.h>
#include <hip/hip_bf16.h>
#include <stdint.h>

#define NDIM 2048
#define MSEQ 8192
#define NG   (3 * NDIM)   // 6144

#define BM 64
#define BN 256
#define BK 32
#define UPB 16            // hidden units per scan block

typedef __bf16 bf16x16 __attribute__((ext_vector_type(16)));
typedef float  f32x8   __attribute__((ext_vector_type(8)));
typedef unsigned int u32x4 __attribute__((ext_vector_type(4)));

union FragBF {
  u32x4   q[2];
  bf16x16 v;
};

__device__ __forceinline__ unsigned short f2bf(float f) {
  unsigned u = __builtin_bit_cast(unsigned, f);
  u += 0x7FFFu + ((u >> 16) & 1u);   // round-to-nearest-even
  return (unsigned short)(u >> 16);
}

__device__ __forceinline__ float sigmoidf(float x) {
  return 1.0f / (1.0f + __expf(-x));
}

// CDNA5 async copy: global -> LDS, 16 bytes per lane, tracked by ASYNCcnt.
__device__ __forceinline__ void async_load16(unsigned lds_off, const void* gaddr) {
  asm volatile("global_load_async_to_lds_b128 %0, %1, off"
               :: "v"(lds_off), "v"(gaddr) : "memory");
}

// ---------------------------------------------------------------- convert x
__global__ __launch_bounds__(256) void k_f32_to_bf16x4(
    const float* __restrict__ in, unsigned short* __restrict__ out, int n4) {
  int i = blockIdx.x * 256 + threadIdx.x;
  if (i >= n4) return;
  const float4 v = reinterpret_cast<const float4*>(in)[i];
  ushort4 o;
  o.x = f2bf(v.x); o.y = f2bf(v.y); o.z = f2bf(v.z); o.w = f2bf(v.w);
  reinterpret_cast<ushort4*>(out)[i] = o;
}

// --------------------------------------------- transpose W (KxN) -> WT (NxK) bf16
__global__ __launch_bounds__(256) void k_transpose_f32_to_bf16(
    const float* __restrict__ in, unsigned short* __restrict__ out,
    int rows /*K*/, int cols /*N*/) {
  __shared__ float tile[32][33];
  const int bx = blockIdx.x * 32, by = blockIdx.y * 32;
  const int tx = threadIdx.x, ty = threadIdx.y;
#pragma unroll
  for (int j = 0; j < 32; j += 8)
    tile[ty + j][tx] = in[(size_t)(by + ty + j) * cols + (bx + tx)];
  __syncthreads();
#pragma unroll
  for (int j = 0; j < 32; j += 8)
    out[(size_t)(bx + ty + j) * rows + (by + tx)] = f2bf(tile[tx][ty + j]);
}

// ---------------------------------------------------- gx = x @ Wx + b (WMMA bf16)
// Block tile 64x256, 8 waves each 32x64 (2x4 wmma tiles).
// Tiles staged with global_load_async_to_lds_b128, double buffered.
__global__ __launch_bounds__(256) void k_gemm_bias(
    const unsigned short* __restrict__ A,   // M x K bf16 (x)
    const unsigned short* __restrict__ BT,  // N x K bf16 (WxT)
    const float* __restrict__ bias,         // N f32
    float* __restrict__ C,                  // M x N f32 (gx)
    int M, int N, int K) {
  __shared__ alignas(16) unsigned short sA[2][BM * BK];   // 2 x 4 KB
  __shared__ alignas(16) unsigned short sB[2][BN * BK];   // 2 x 16 KB (N-major)

  const int tid  = threadIdx.x;
  const int lane = tid & 31;
  const int wave = tid >> 5;
  const int wm   = wave >> 2;     // 0..1
  const int wn   = wave & 3;      // 0..3
  const int half = lane >> 4;
  const int ln   = lane & 15;

  const int m0 = blockIdx.y * BM;
  const int n0 = blockIdx.x * BN;

  f32x8 acc[2][4] = {};

  const unsigned ldsA[2] = {(unsigned)(size_t)&sA[0][0], (unsigned)(size_t)&sA[1][0]};
  const unsigned ldsB[2] = {(unsigned)(size_t)&sB[0][0], (unsigned)(size_t)&sB[1][0]};

  const int NK = K / BK;

  // stage(kc, buf): 1 A chunk + 4 B chunks per thread = 5 async ops per wave
  auto stage = [&](int kb, int buf) {
    {
      const int row = tid >> 2, chk = tid & 3;
      async_load16(ldsA[buf] + (unsigned)tid * 16u,
                   A + (size_t)(m0 + row) * K + kb * BK + chk * 8);
    }
#pragma unroll
    for (int j = 0; j < 4; ++j) {
      const int idx = tid + j * 256;
      const int row = idx >> 2, chk = idx & 3;
      async_load16(ldsB[buf] + (unsigned)idx * 16u,
                   BT + (size_t)(n0 + row) * K + kb * BK + chk * 8);
    }
  };

  stage(0, 0);

  for (int k = 0; k < NK; ++k) {
    const int buf = k & 1;
    if (k + 1 < NK) {
      stage(k + 1, buf ^ 1);
      // async loads retire in order: <=5 outstanding means stage k landed
      asm volatile("s_wait_asynccnt 0x5" ::: "memory");
    } else {
      asm volatile("s_wait_asynccnt 0x0" ::: "memory");
    }
    __syncthreads();

    const unsigned short* bufA = sA[buf];
    const unsigned short* bufB = sB[buf];

    FragBF fa[2], fb[4];
#pragma unroll
    for (int mi = 0; mi < 2; ++mi) {
      const unsigned short* p = bufA + (wm * 32 + mi * 16 + ln) * BK;
      fa[mi].q[0] = *reinterpret_cast<const u32x4*>(p + 16 * half);
      fa[mi].q[1] = *reinterpret_cast<const u32x4*>(p + 16 * half + 8);
    }
#pragma unroll
    for (int ni = 0; ni < 4; ++ni) {
      const unsigned short* p = bufB + (wn * 64 + ni * 16 + ln) * BK + 16 * half;
      fb[ni].q[0] = *reinterpret_cast<const u32x4*>(p);
      fb[ni].q[1] = *reinterpret_cast<const u32x4*>(p + 8);
    }
#pragma unroll
    for (int mi = 0; mi < 2; ++mi)
#pragma unroll
      for (int ni = 0; ni < 4; ++ni)
        acc[mi][ni] = __builtin_amdgcn_wmma_f32_16x16x32_bf16(
            false, fa[mi].v, false, fb[ni].v, (short)0, acc[mi][ni], false, false);
    __syncthreads();
  }

  // epilogue: C = acc + bias (C/D layout: lane half->M+=8, vgpr r -> M=r)
#pragma unroll
  for (int mi = 0; mi < 2; ++mi) {
#pragma unroll
    for (int ni = 0; ni < 4; ++ni) {
      const int gcol  = n0 + wn * 64 + ni * 16 + ln;
      const float bv  = bias[gcol];
      const int grow0 = m0 + wm * 32 + mi * 16 + 8 * half;
#pragma unroll
      for (int r = 0; r < 8; ++r)
        C[(size_t)(grow0 + r) * N + gcol] = acc[mi][ni][r] + bv;
    }
  }
}

// ------------------------------------------------------------ init h + barrier
__global__ void k_init(unsigned short* h, unsigned* syn) {
  const int tid = threadIdx.x;
  for (int i = tid; i < 2 * NDIM; i += 256) h[i] = 0;
  if (tid < 64) syn[tid] = 0;
}

// ------------------------------------------------- persistent GRU scan kernel
__global__ __launch_bounds__(256) void k_gru_scan(
    const unsigned short* __restrict__ WhT,  // NG x NDIM bf16 (L2-resident)
    const float* __restrict__ gx,            // MSEQ x NG f32
    unsigned short* __restrict__ hbuf,       // 2 x NDIM bf16 (ping-pong)
    float* __restrict__ out,                 // MSEQ x NDIM f32
    unsigned* __restrict__ syn) {
  const int tid  = threadIdx.x;
  const int lane = tid & 31;
  const int wave = tid >> 5;
  const int half = lane >> 4;
  const int ln   = lane & 15;
  const int u0   = blockIdx.x * UPB;

  __shared__ float partials[8][3][16];
  __shared__ float gates[3][16];

  const unsigned short* wz = WhT + (size_t)(u0 + ln) * NDIM;
  const unsigned short* wr = WhT + (size_t)(NDIM + u0 + ln) * NDIM;
  const unsigned short* wn = WhT + (size_t)(2 * NDIM + u0 + ln) * NDIM;

  float hprev = 0.0f;   // owned unit state (tid < UPB)

  for (int t = 0; t < MSEQ; ++t) {
    const unsigned short* hrd = hbuf + (t & 1) * NDIM;
    unsigned short*       hwr = hbuf + ((t + 1) & 1) * NDIM;

    f32x8 accz = {}, accr = {}, accn = {};
#pragma unroll
    for (int i = 0; i < 8; ++i) {
      const int kc = (i * 8 + wave) * BK;
      FragBF fa, fz, fr, fn;
      // broadcast-A: every row of A = h[kc..kc+31]
      fa.q[0] = *reinterpret_cast<const u32x4*>(hrd + kc + 16 * half);
      fa.q[1] = *reinterpret_cast<const u32x4*>(hrd + kc + 16 * half + 8);
      const int off = kc + 16 * half;
      fz.q[0] = *reinterpret_cast<const u32x4*>(wz + off);
      fz.q[1] = *reinterpret_cast<const u32x4*>(wz + off + 8);
      fr.q[0] = *reinterpret_cast<const u32x4*>(wr + off);
      fr.q[1] = *reinterpret_cast<const u32x4*>(wr + off + 8);
      fn.q[0] = *reinterpret_cast<const u32x4*>(wn + off);
      fn.q[1] = *reinterpret_cast<const u32x4*>(wn + off + 8);
      accz = __builtin_amdgcn_wmma_f32_16x16x32_bf16(false, fa.v, false, fz.v, (short)0, accz, false, false);
      accr = __builtin_amdgcn_wmma_f32_16x16x32_bf16(false, fa.v, false, fr.v, (short)0, accr, false, false);
      accn = __builtin_amdgcn_wmma_f32_16x16x32_bf16(false, fa.v, false, fn.v, (short)0, accn, false, false);
    }
    if (half == 0) {   // D row M=0 lives in vgpr0 of lanes 0..15
      partials[wave][0][ln] = accz[0];
      partials[wave][1][ln] = accr[0];
      partials[wave][2][ln] = accn[0];
    }
    __syncthreads();
    if (tid < 48) {
      const int g = tid >> 4, nn = tid & 15;
      float s = 0.f;
#pragma unroll
      for (int w = 0; w < 8; ++w) s += partials[w][g][nn];
      gates[g][nn] = s;
    }
    __syncthreads();
    if (tid < UPB) {
      const int u = u0 + tid;
      const float* gxt = gx + (size_t)t * NG;
      const float z = sigmoidf(gxt[u] + gates[0][tid]);
      const float r = sigmoidf(gxt[NDIM + u] + gates[1][tid]);
      const float c = tanhf(gxt[2 * NDIM + u] + r * gates[2][tid]);
      const float hn = (1.0f - z) * hprev + z * c;
      hprev = hn;
      out[(size_t)t * NDIM + u] = hn;
      hwr[u] = f2bf(hn);
      if (tid == 0 && t + 8 < MSEQ) {   // pull future gx rows toward L2
        const float* pf = gx + (size_t)(t + 8) * NG;
        __builtin_prefetch(pf + u0, 0, 0);
        __builtin_prefetch(pf + NDIM + u0, 0, 0);
        __builtin_prefetch(pf + 2 * NDIM + u0, 0, 0);
      }
    }
    __syncthreads();
    // ---- device-wide barrier: all h writes visible before next step ----
    if (tid == 0) {
      __threadfence();
      const unsigned prev = __hip_atomic_fetch_add(&syn[0], 1u, __ATOMIC_ACQ_REL,
                                                   __HIP_MEMORY_SCOPE_AGENT);
      if (prev == (unsigned)(gridDim.x - 1)) {
        __hip_atomic_store(&syn[0], 0u, __ATOMIC_RELAXED, __HIP_MEMORY_SCOPE_AGENT);
        __hip_atomic_store(&syn[1], (unsigned)(t + 1), __ATOMIC_RELEASE,
                           __HIP_MEMORY_SCOPE_AGENT);
      } else {
        while (__hip_atomic_load(&syn[1], __ATOMIC_ACQUIRE,
                                 __HIP_MEMORY_SCOPE_AGENT) <= (unsigned)t)
          __builtin_amdgcn_s_sleep(2);
      }
    }
    __syncthreads();
  }
}

// ---------------------------------------------------------------- launcher
extern "C" void kernel_launch(void* const* d_in, const int* in_sizes, int n_in,
                              void* d_out, int out_size, void* d_ws, size_t ws_size,
                              hipStream_t stream) {
  const float* x  = (const float*)d_in[0];
  const float* Wx = (const float*)d_in[1];
  const float* Wh = (const float*)d_in[2];
  const float* b  = (const float*)d_in[3];
  float* out = (float*)d_out;

  char* ws = (char*)d_ws;
  // workspace layout (all 256B-aligned):
  unsigned short* x_bf = (unsigned short*)(ws + 0);           //  33,554,432 B
  unsigned short* WxT  = (unsigned short*)(ws + 33554432);    //  25,165,824 B
  unsigned short* WhT  = (unsigned short*)(ws + 58720256);    //  25,165,824 B
  float*          gx   = (float*)(ws + 83886080);             // 201,326,592 B
  unsigned short* hbuf = (unsigned short*)(ws + 285212672);   //       8,192 B
  unsigned*       syn  = (unsigned*)(ws + 285220864);         //         256 B

  {   // x -> bf16
    const int n4 = MSEQ * NDIM / 4;
    k_f32_to_bf16x4<<<n4 / 256, 256, 0, stream>>>(x, x_bf, n4);
  }
  {   // Wx, Wh -> transposed bf16 (N-major)
    dim3 blk(32, 8), grd(NG / 32, NDIM / 32);
    k_transpose_f32_to_bf16<<<grd, blk, 0, stream>>>(Wx, WxT, NDIM, NG);
    k_transpose_f32_to_bf16<<<grd, blk, 0, stream>>>(Wh, WhT, NDIM, NG);
  }
  {   // gx = x @ Wx + b
    dim3 grd(NG / BN, MSEQ / BM);
    k_gemm_bias<<<grd, 256, 0, stream>>>(x_bf, WxT, b, gx, MSEQ, NG, NDIM);
  }
  k_init<<<1, 256, 0, stream>>>(hbuf, syn);
  k_gru_scan<<<NDIM / UPB, 256, 0, stream>>>(WhT, gx, hbuf, out, syn);
}
